// GATModel_67095979099185
// MI455X (gfx1250) — compile-verified, matrix operand order checked by reading, not run on hardware
//
#include <hip/hip_runtime.h>
#include <hip/hip_bf16.h>

// ---------------------------------------------------------------------------
// Types for CDNA5 WMMA (wave32): 16x16x32 bf16 -> f32 accumulate
// ---------------------------------------------------------------------------
typedef __attribute__((ext_vector_type(16))) __bf16 v16bf;
typedef __attribute__((ext_vector_type(8)))  __bf16 v8bf;
typedef __attribute__((ext_vector_type(8)))  float  v8f;

#define NEG_SLOPE 0.2f

__device__ __forceinline__ float leaky(float v) {
    return v > 0.f ? v : NEG_SLOPE * v;
}

// Ordered-int trick for float atomic max (init value must be -inf).
__device__ __forceinline__ void atomicMaxF(float* addr, float val) {
    if (val >= 0.f)
        atomicMax((int*)addr, __float_as_int(val));
    else
        atomicMin((unsigned int*)addr, (unsigned int)__float_as_int(val));
}

// ---------------------------------------------------------------------------
// f32 -> bf16 conversion kernels (stage GEMM operands in bf16; W transposed
// so the B-fragment load mirrors the A-fragment load pattern)
// ---------------------------------------------------------------------------
__global__ void cvt_bf16_kernel(const float* __restrict__ src,
                                __bf16* __restrict__ dst, long n) {
    long i = (long)blockIdx.x * blockDim.x + threadIdx.x;
    if (i < n) dst[i] = (__bf16)src[i];
}

// W: [K, Ncols] row-major f32  ->  Wt: [Ncols, K] row-major bf16
__global__ void cvt_bf16_transpose_kernel(const float* __restrict__ W,
                                          __bf16* __restrict__ Wt,
                                          int K, int Ncols) {
    int i = blockIdx.x * blockDim.x + threadIdx.x;
    if (i >= K * Ncols) return;
    int k = i / Ncols;
    int c = i - k * Ncols;
    Wt[(long)c * K + k] = (__bf16)W[i];
}

// ---------------------------------------------------------------------------
// WMMA GEMM: C[M,Ncols] = A[M,K] * B[K,Ncols], B supplied transposed (Bt[N,K]).
// One wave computes one 16x16 f32 tile; K stepped by 32 with
// v_wmma_f32_16x16x32_bf16. Requires M%16==0, K%32==0, Ncols%16==0.
//
// A-fragment (16-bit A 16x32, ISA 7.12.2): lane half 0 holds K=0..7,16..23,
// half 1 holds K=8..15,24..31 -> two contiguous 8-element (16B) loads.
// B-fragment uses the mirrored layout (lane = column) on the transposed B.
// C/D layout: VGPR r, lane l -> (M = r + 8*(l>>4), N = l&15).
// ---------------------------------------------------------------------------
__global__ void wmma_gemm_bf16(const __bf16* __restrict__ A,
                               const __bf16* __restrict__ Bt,
                               float* __restrict__ C,
                               int M, int K, int Ncols) {
    int wid  = (int)((blockIdx.x * (long)blockDim.x + threadIdx.x) >> 5);
    int lane = threadIdx.x & 31;
    int ntn  = Ncols >> 4;
    int tile_m = wid / ntn;
    int tile_n = wid - tile_m * ntn;
    if (tile_m * 16 >= M) return;                 // wave-uniform

    int half = lane >> 4;                          // 0 or 1
    int idx  = lane & 15;                          // row (A) / col (B)

    const __bf16* arow = A  + (size_t)(tile_m * 16 + idx) * K;
    const __bf16* brow = Bt + (size_t)(tile_n * 16 + idx) * K;

    v8f acc = {};
    for (int k0 = 0; k0 < K; k0 += 32) {
        v8bf alo = *(const v8bf*)(arow + k0 +  8 * half);
        v8bf ahi = *(const v8bf*)(arow + k0 + 16 + 8 * half);
        v8bf blo = *(const v8bf*)(brow + k0 +  8 * half);
        v8bf bhi = *(const v8bf*)(brow + k0 + 16 + 8 * half);
        if (k0 + 32 < K) {                         // speculative L2 prefetch
            __builtin_prefetch(arow + k0 + 32, 0, 1);
            __builtin_prefetch(brow + k0 + 32, 0, 1);
        }
        v16bf a = __builtin_shufflevector(alo, ahi,
                    0,1,2,3,4,5,6,7,8,9,10,11,12,13,14,15);
        v16bf b = __builtin_shufflevector(blo, bhi,
                    0,1,2,3,4,5,6,7,8,9,10,11,12,13,14,15);
        acc = __builtin_amdgcn_wmma_f32_16x16x32_bf16(
                  false, a, false, b, (short)0, acc, false, false);
    }

    float* crow = C + (size_t)(tile_m * 16 + 8 * half) * Ncols + tile_n * 16 + idx;
#pragma unroll
    for (int r = 0; r < 8; ++r)
        crow[(size_t)r * Ncols] = acc[r];
}

// ---------------------------------------------------------------------------
// Per-(node,head) attention scores: s = <h_row, a_src>, t = <h_row, a_dst>.
// One wave per row of length C; also initializes segment-max / denom.
// h is [N, H*C] flat, so row index gw = node*H + head addresses h + gw*C.
// ---------------------------------------------------------------------------
__global__ void scores_kernel(const float* __restrict__ hfeat,
                              const float* __restrict__ a_src,
                              const float* __restrict__ a_dst,
                              int NH, int H, int C,
                              float* __restrict__ s, float* __restrict__ t,
                              float* __restrict__ m, float* __restrict__ den) {
    int gw   = (int)((blockIdx.x * (long)blockDim.x + threadIdx.x) >> 5);
    int lane = threadIdx.x & 31;
    if (gw >= NH) return;                          // wave-uniform
    int head = gw % H;
    const float* row = hfeat + (long)gw * C;
    const float* as  = a_src + head * C;
    const float* ad  = a_dst + head * C;
    float accs = 0.f, accd = 0.f;
    for (int i = lane; i < C; i += 32) {
        float v = row[i];
        accs += v * as[i];
        accd += v * ad[i];
    }
#pragma unroll
    for (int off = 16; off > 0; off >>= 1) {
        accs += __shfl_xor(accs, off, 32);
        accd += __shfl_xor(accd, off, 32);
    }
    if (lane == 0) {
        s[gw]   = accs;
        t[gw]   = accd;
        m[gw]   = -__builtin_inff();
        den[gw] = 0.f;
    }
}

// ---------------------------------------------------------------------------
// Edge passes. Edge e < E comes from edge_index (row 0 = src, row 1 = dst);
// e >= E are the self-loops (src = dst = e - E).
// ---------------------------------------------------------------------------
template <int H>
__global__ void edge_max_kernel(const int* __restrict__ ei, int E, int N,
                                const float* __restrict__ s,
                                const float* __restrict__ t,
                                float* __restrict__ m) {
    int e = (int)(blockIdx.x * (long)blockDim.x + threadIdx.x);
    if (e >= E + N) return;
    int src = (e < E) ? ei[e]     : (e - E);
    int dst = (e < E) ? ei[E + e] : (e - E);
#pragma unroll
    for (int h = 0; h < H; ++h)
        atomicMaxF(&m[dst * H + h], leaky(s[src * H + h] + t[dst * H + h]));
}

template <int H>
__global__ void edge_sum_kernel(const int* __restrict__ ei, int E, int N,
                                const float* __restrict__ s,
                                const float* __restrict__ t,
                                const float* __restrict__ m,
                                float* __restrict__ den) {
    int e = (int)(blockIdx.x * (long)blockDim.x + threadIdx.x);
    if (e >= E + N) return;
    int src = (e < E) ? ei[e]     : (e - E);
    int dst = (e < E) ? ei[E + e] : (e - E);
#pragma unroll
    for (int h = 0; h < H; ++h) {
        float v = leaky(s[src * H + h] + t[dst * H + h]);
        atomicAdd(&den[dst * H + h], __expf(v - m[dst * H + h]));
    }
}

// One block per edge, one lane per feature channel: agg[dst] += alpha*h[src].
template <int H, int C>
__global__ void edge_agg_kernel(const int* __restrict__ ei, int E, int N,
                                const float* __restrict__ s,
                                const float* __restrict__ t,
                                const float* __restrict__ m,
                                const float* __restrict__ den,
                                const float* __restrict__ hf,
                                float* __restrict__ agg) {
    constexpr int HC = H * C;
    int e = blockIdx.x;
    int c = threadIdx.x;                            // 0..HC-1
    int h = c / C;
    int src = (e < E) ? ei[e]     : (e - E);
    int dst = (e < E) ? ei[E + e] : (e - E);
    float v     = leaky(s[src * H + h] + t[dst * H + h]);
    float alpha = __expf(v - m[dst * H + h]) / den[dst * H + h];
    atomicAdd(&agg[(long)dst * HC + c], alpha * hf[(long)src * HC + c]);
}

// Layer-1 epilogue: h1act = elu(agg + b1), emitted in bf16 for GEMM2.
__global__ void elu_bias_cvt_kernel(const float* __restrict__ agg,
                                    const float* __restrict__ b,
                                    __bf16* __restrict__ dst, long n, int HC) {
    long i = (long)blockIdx.x * blockDim.x + threadIdx.x;
    if (i >= n) return;
    float v = agg[i] + b[(int)(i % HC)];
    v = v > 0.f ? v : (__expf(v) - 1.f);
    dst[i] = (__bf16)v;
}

__global__ void out_bias_kernel(const float* __restrict__ agg,
                                const float* __restrict__ b,
                                float* __restrict__ out, long n, int C) {
    long i = (long)blockIdx.x * blockDim.x + threadIdx.x;
    if (i >= n) return;
    out[i] = agg[i] + b[(int)(i % C)];
}

// ---------------------------------------------------------------------------
// Host launcher
// ---------------------------------------------------------------------------
extern "C" void kernel_launch(void* const* d_in, const int* in_sizes, int n_in,
                              void* d_out, int out_size, void* d_ws, size_t ws_size,
                              hipStream_t stream) {
    (void)n_in; (void)out_size; (void)ws_size;

    const float* x   = (const float*)d_in[0];
    const int*   ei  = (const int*)  d_in[1];
    const float* W1  = (const float*)d_in[2];
    const float* as1 = (const float*)d_in[3];
    const float* ad1 = (const float*)d_in[4];
    const float* b1  = (const float*)d_in[5];
    const float* W2  = (const float*)d_in[6];
    const float* as2 = (const float*)d_in[7];
    const float* ad2 = (const float*)d_in[8];
    const float* b2  = (const float*)d_in[9];

    const int F_IN = 128, HEADS = 4, HID = 64, HC1 = HEADS * HID /*256*/, F_OUT = 128;
    const int N  = in_sizes[0] / F_IN;      // 50000
    const int E  = in_sizes[1] / 2;         // 800000
    const int ET = E + N;

    // Workspace carve-out (256B aligned chunks); later stages alias dead buffers.
    size_t off = 0;
    auto alloc = [&](size_t bytes) -> void* {
        void* p = (char*)d_ws + off;
        off += (bytes + 255) & ~(size_t)255;
        return p;
    };
    __bf16* x_bf  = (__bf16*)alloc((size_t)N * F_IN * 2);        // 12.8 MB
    __bf16* W1t   = (__bf16*)alloc((size_t)HC1 * F_IN * 2);      // W1^T [256,128]
    __bf16* W2t   = (__bf16*)alloc((size_t)F_OUT * HC1 * 2);     // W2^T [128,256]
    float*  h1    = (float*) alloc((size_t)N * HC1 * 4);         // 51.2 MB (reused as h2)
    float*  sb    = (float*) alloc((size_t)N * HEADS * 4);       // scores (reused L2)
    float*  tb    = (float*) alloc((size_t)N * HEADS * 4);
    float*  mb    = (float*) alloc((size_t)N * HEADS * 4);
    float*  db    = (float*) alloc((size_t)N * HEADS * 4);
    float*  agg   = (float*) alloc((size_t)N * HC1 * 4);         // 51.2 MB (reused L2)
    __bf16* h1abf = (__bf16*)alloc((size_t)N * HC1 * 2);         // 25.6 MB
    float*  h2    = h1;                                          // alias (dead after L1)
    float*  agg2  = agg;                                         // alias (dead after elu)

    // --- stage operands in bf16 ------------------------------------------
    long nx = (long)N * F_IN;
    cvt_bf16_kernel<<<(int)((nx + 255) / 256), 256, 0, stream>>>(x, x_bf, nx);
    cvt_bf16_transpose_kernel<<<(F_IN * HC1 + 255) / 256, 256, 0, stream>>>(W1, W1t, F_IN, HC1);
    cvt_bf16_transpose_kernel<<<(HC1 * F_OUT + 255) / 256, 256, 0, stream>>>(W2, W2t, HC1, F_OUT);

    // --- layer 1 ----------------------------------------------------------
    {
        int waves = (N / 16) * (HC1 / 16);
        wmma_gemm_bf16<<<(waves * 32 + 255) / 256, 256, 0, stream>>>(x_bf, W1t, h1, N, F_IN, HC1);
    }
    {
        int NH = N * HEADS;
        scores_kernel<<<(NH * 32 + 255) / 256, 256, 0, stream>>>(h1, as1, ad1, NH, HEADS, HID,
                                                                 sb, tb, mb, db);
    }
    hipMemsetAsync(agg, 0, (size_t)N * HC1 * 4, stream);
    edge_max_kernel<4><<<(ET + 255) / 256, 256, 0, stream>>>(ei, E, N, sb, tb, mb);
    edge_sum_kernel<4><<<(ET + 255) / 256, 256, 0, stream>>>(ei, E, N, sb, tb, mb, db);
    edge_agg_kernel<4, 64><<<ET, 256, 0, stream>>>(ei, E, N, sb, tb, mb, db, h1, agg);

    long n1 = (long)N * HC1;
    elu_bias_cvt_kernel<<<(int)((n1 + 255) / 256), 256, 0, stream>>>(agg, b1, h1abf, n1, HC1);

    // --- layer 2 ----------------------------------------------------------
    {
        int waves = (N / 16) * (F_OUT / 16);
        wmma_gemm_bf16<<<(waves * 32 + 255) / 256, 256, 0, stream>>>(h1abf, W2t, h2, N, HC1, F_OUT);
    }
    scores_kernel<<<(N * 32 + 255) / 256, 256, 0, stream>>>(h2, as2, ad2, N, 1, F_OUT,
                                                            sb, tb, mb, db);
    hipMemsetAsync(agg2, 0, (size_t)N * F_OUT * 4, stream);
    edge_max_kernel<1><<<(ET + 255) / 256, 256, 0, stream>>>(ei, E, N, sb, tb, mb);
    edge_sum_kernel<1><<<(ET + 255) / 256, 256, 0, stream>>>(ei, E, N, sb, tb, mb, db);
    edge_agg_kernel<1, 128><<<ET, 128, 0, stream>>>(ei, E, N, sb, tb, mb, db, h2, agg2);

    long n2 = (long)N * F_OUT;
    out_bias_kernel<<<(int)((n2 + 255) / 256), 256, 0, stream>>>(agg2, b2, (float*)d_out, n2, F_OUT);
}
